// MoEFFN_5738076307767
// MI455X (gfx1250) — matrix operand order repaired and checked by reading, not compiled, not run on hardware
//
#include <hip/hip_runtime.h>
#include <hip/hip_bf16.h>
#include <math.h>

// MoE FFN for MI455X (gfx1250, wave32, WMMA + TDM).
//   t[N,E*R]   = x[N,H] @ W1cat[H,E*R], scaled by top-2 routing weights
//   hb[N,F]    = gelu(t @ W2flat[E*R,F])
//   out[N,H]   = hb @ lin2_w[F,H] + lin2_b

typedef __attribute__((ext_vector_type(16))) _Float16 v16h;
typedef __attribute__((ext_vector_type(8)))  float    v8f;
typedef __attribute__((ext_vector_type(4)))  unsigned int v4u;
typedef __attribute__((ext_vector_type(8)))  int      v8i;
typedef __attribute__((ext_vector_type(4)))  int      v4i;

#define NTOK 8192   // B*S
#define HDIM 1024
#define RDIM 256
#define FDIM 4096
#define NEXP 8
#define ERD  2048   // E*R

#if __has_builtin(__builtin_amdgcn_tensor_load_to_lds) && \
    __has_builtin(__builtin_amdgcn_s_wait_tensorcnt)
#define USE_TDM 1
#else
#define USE_TDM 0
#endif

// ---------------------------------------------------------------- router ----
__global__ __launch_bounds__(256)
void router_kernel(const float* __restrict__ x, const float* __restrict__ rw,
                   const float* __restrict__ rb, float* __restrict__ mw)
{
    int lane = threadIdx.x & 31;
    int wid  = threadIdx.x >> 5;
    int tok  = blockIdx.x * 8 + wid;
    if (tok >= NTOK) return;

    const float* xr = x + (size_t)tok * HDIM;
    float acc[NEXP];
#pragma unroll
    for (int e = 0; e < NEXP; ++e) acc[e] = 0.f;

    for (int h = lane; h < HDIM; h += 32) {
        float xv = xr[h];
        const float* wp = rw + (size_t)h * NEXP;   // router_w [H,E], row-contig
#pragma unroll
        for (int e = 0; e < NEXP; ++e) acc[e] += xv * wp[e];
    }
#pragma unroll
    for (int e = 0; e < NEXP; ++e) {
#pragma unroll
        for (int off = 16; off > 0; off >>= 1)
            acc[e] += __shfl_xor(acc[e], off, 32);
    }
    if (lane == 0) {
        float l[NEXP], p[NEXP];
        float mx = -3.0e38f;
#pragma unroll
        for (int e = 0; e < NEXP; ++e) { l[e] = acc[e] + rb[e]; mx = fmaxf(mx, l[e]); }
        float s = 0.f;
#pragma unroll
        for (int e = 0; e < NEXP; ++e) { p[e] = __expf(l[e] - mx); s += p[e]; }
        float inv = 1.f / s;
#pragma unroll
        for (int e = 0; e < NEXP; ++e) p[e] *= inv;
        int i1 = 0;
#pragma unroll
        for (int e = 1; e < NEXP; ++e) if (p[e] > p[i1]) i1 = e;
        int i2 = (i1 == 0) ? 1 : 0;
#pragma unroll
        for (int e = 0; e < NEXP; ++e) if (e != i1 && p[e] > p[i2]) i2 = e;
#pragma unroll
        for (int e = 0; e < NEXP; ++e)
            mw[(size_t)tok * NEXP + e] = (e == i1 || e == i2) ? p[e] : 0.f;
    }
}

// ------------------------------------------------------------- converts ----
__global__ void cvt_f32_f16(const float* __restrict__ in, _Float16* __restrict__ out, long n)
{
    long i = (long)blockIdx.x * blockDim.x + threadIdx.x;
    long stride = (long)gridDim.x * blockDim.x;
    for (; i < n; i += stride) out[i] = (_Float16)in[i];
}

// w1 [E,H,R] -> W1cat f16 [H, E*R] so GEMM1's B is plain row-major
__global__ void pack_w1_kernel(const float* __restrict__ w1, _Float16* __restrict__ o)
{
    long n = (long)HDIM * ERD;
    long i = (long)blockIdx.x * blockDim.x + threadIdx.x;
    long stride = (long)gridDim.x * blockDim.x;
    for (; i < n; i += stride) {
        int h = (int)(i >> 11);         // ERD = 2048
        int c = (int)(i & 2047);
        int e = c >> 8;                 // RDIM = 256
        int r = c & 255;
        o[i] = (_Float16)w1[((size_t)e * HDIM + h) * RDIM + r];
    }
}

// ------------------------------------------------------------- TDM issue ----
#if USE_TDM
// Build a 2D D# (ISA ch.8) for a 128x32 f16 activation tile and issue
// TENSOR_LOAD_TO_LDS. tile_dim0=32 (contig k), tile_dim1=128 (rows, stride=lda).
// pad_enable: interval = 16 dwords (one 32-f16 row), amount = 4 dwords
// -> reproduces the stride-40 padded LDS rows the WMMA fragment loads expect.
__device__ __forceinline__ void tdm_load_a(const _Float16* gsrc, unsigned lds_off, int lda)
{
    unsigned long long ga = (unsigned long long)(size_t)gsrc;
    v4u g0;
    g0.x = 1u;                                          // count=1, user mode
    g0.y = lds_off;                                     // lds_addr
    g0.z = (unsigned)(ga & 0xffffffffu);                // global_addr[31:0]
    g0.w = (unsigned)((ga >> 32) & 0x1ffffffu) | (2u << 30);  // addr[56:32] | type=2

    unsigned td0 = (unsigned)lda;                       // tensor_dim0 (elements)
    unsigned td1 = 1u << 20;                            // plenty of rows (no OOB clip)
    unsigned st0 = (unsigned)lda;                       // dim0 stride (elements)
    v8i g1;
    g1[0] = (int)((1u << 16)      // data_size = 2 bytes
                | (1u << 20)      // pad_enable
                | (3u << 22)      // pad_interval: 16 dwords
                | (3u << 25));    // pad_amount: 4 dwords
    g1[1] = (int)((td0 & 0xffffu) << 16);               // [15:0] atomic addr = 0
    g1[2] = (int)((td0 >> 16) | ((td1 & 0xffffu) << 16));
    g1[3] = (int)((td1 >> 16) | (32u << 16));           // tile_dim0 = 32
    g1[4] = (int)128u;                                  // tile_dim1 = 128, tile_dim2 = 0
    g1[5] = (int)st0;                                   // dim0_stride[31:0]
    g1[6] = 0;                                          // stride hi / dim1 stride
    g1[7] = 0;
    v4i z4 = {0, 0, 0, 0};                              // groups 2/3 unused (2D)
    v8i z8 = {0, 0, 0, 0, 0, 0, 0, 0};
    __builtin_amdgcn_tensor_load_to_lds(g0, g1, z4, z4, z8, 0);
}
#endif

// ---------------------------------------------------------------- GEMM -----
// C[M,N] = A[M,K](f16 rowmajor) x B[K,N](f16 rowmajor), f32 accum, WMMA 16x16x32.
// 128x128 block tile, 256 threads = 8 waves; wave (wm,wn) owns 64x32 = 4x2 tiles.
// Double-buffered LDS: A tiles via TDM (TENSORcnt pipeline, depth 2) when
// available, B tiles via register-lookahead software pipeline (transpose-packed
// into LDS as (k,k+1) dword pairs).
// EPI 0: *= mw[row, col/256] f16 | EPI 1: erf-gelu f16 | EPI 2: +bias f32
template<int EPI>
__global__ __launch_bounds__(256)
void gemm_wmma_f16(const _Float16* __restrict__ A, const _Float16* __restrict__ B,
                   int Kd, int lda, int ldb, int ldc,
                   const float* __restrict__ extra,
                   _Float16* __restrict__ outh, float* __restrict__ outf)
{
    __shared__ _Float16 lA[2][128][40];   // [m][k] + pad
    __shared__ _Float16 lB[2][128][40];   // transposed [n][k] + pad

    const int tid  = threadIdx.x;
    const int lane = tid & 31;
    const int w    = tid >> 5;
    const int wm   = w & 1;
    const int wn   = w >> 1;
    const int l15  = lane & 15;
    const int lhi  = lane >> 4;

    const int n0 = blockIdx.x * 128;
    const int m0 = blockIdx.y * 128;

    // B staging: thread covers 8 cols x 2 k-rows (two 16B loads, 8 dword stores)
    const int bn8 = (tid & 15) << 3;      // 0..120 step 8
    const int bkp = (tid >> 4) << 1;      // 0..30 step 2
    const _Float16* gB = B + (size_t)bkp * ldb + n0 + bn8;

#if !USE_TDM
    const int ar = tid >> 1;              // A row 0..127
    const int ac = (tid & 1) << 4;        // A col 0 or 16
    const _Float16* gA = A + (size_t)(m0 + ar) * lda + ac;
#endif

    v8f c[4][2];
#pragma unroll
    for (int i = 0; i < 4; ++i)
#pragma unroll
        for (int j = 0; j < 2; ++j)
            c[i][j] = (v8f){0.f, 0.f, 0.f, 0.f, 0.f, 0.f, 0.f, 0.f};

    // ---- prologue: put tile 0 in flight
    union BQ { uint4 q; unsigned short u[8]; };
    BQ qb0, qb1;
    qb0.q = *(const uint4*)gB;
    qb1.q = *(const uint4*)(gB + ldb);
#if USE_TDM
    if (w == 0)
        tdm_load_a(A + (size_t)m0 * lda, (unsigned)(size_t)&lA[0][0][0], lda);
#else
    uint4 qa0 = ((const uint4*)gA)[0];
    uint4 qa1 = ((const uint4*)gA)[1];
#endif

    for (int k0 = 0; k0 < Kd; k0 += 32) {
        const int  s    = (k0 >> 5) & 1;
        const bool more = (k0 + 32) < Kd;

        __syncthreads();                 // previous compute done; buffers free

#if USE_TDM
        if (w == 0 && more)              // DMA next A tile into the other buffer
            tdm_load_a(A + (size_t)m0 * lda + (k0 + 32),
                       (unsigned)(size_t)&lA[1 - s][0][0], lda);
#endif
        // store current B tile: packed (k,k+1) pairs, contiguous per n
#pragma unroll
        for (int j = 0; j < 8; ++j) {
            unsigned pak = (unsigned)qb0.u[j] | ((unsigned)qb1.u[j] << 16);
            *(unsigned*)&lB[s][bn8 + j][bkp] = pak;
        }
#if !USE_TDM
        {   uint4* d = (uint4*)&lA[s][ar][ac];
            d[0] = qa0; d[1] = qa1; }
#endif
        if (more) {                       // issue next-tile global loads now
            const _Float16* gBn = gB + (size_t)(k0 + 32) * ldb;
            qb0.q = *(const uint4*)gBn;
            qb1.q = *(const uint4*)(gBn + ldb);
#if !USE_TDM
            const _Float16* gAn = gA + (k0 + 32);
            qa0 = ((const uint4*)gAn)[0];
            qa1 = ((const uint4*)gAn)[1];
#endif
        }
#if USE_TDM
        if (more) __builtin_amdgcn_s_wait_tensorcnt(1);   // tile s landed
        else      __builtin_amdgcn_s_wait_tensorcnt(0);
#endif
        __syncthreads();                 // LDS tile s visible to all waves

        // Fragments in contiguous k order (k = 16*(lane/16) + s) for A and B; both A and B resolve to 2x ds_load_b128.
        v16h afr[4], bfr[2];
#pragma unroll
        for (int mi = 0; mi < 4; ++mi) {
            int r = wm * 64 + mi * 16 + l15;
            union { v16h v; unsigned u[8]; } ua;
#pragma unroll
            for (int p2 = 0; p2 < 8; ++p2) {
                int k = 2 * p2 + lhi * 16;
                ua.u[p2] = *(const unsigned*)&lA[s][r][k];
            }
            afr[mi] = ua.v;
        }
#pragma unroll
        for (int ni = 0; ni < 2; ++ni) {
            int n = wn * 32 + ni * 16 + l15;
            union { v16h v; uint4 q[2]; } ub;
            const uint4* p = (const uint4*)&lB[s][n][lhi * 16];
            ub.q[0] = p[0]; ub.q[1] = p[1];
            bfr[ni] = ub.v;
        }
#pragma unroll
        for (int mi = 0; mi < 4; ++mi)
#pragma unroll
            for (int ni = 0; ni < 2; ++ni)
                c[mi][ni] = __builtin_amdgcn_wmma_f32_16x16x32_f16(
                    false, afr[mi], false, bfr[ni], (short)0, c[mi][ni],
                    false, false);
    }

    // Epilogue. C/D layout: VGPR v -> row = v (+8 for lanes>=16), col = lane&15.
#pragma unroll
    for (int mi = 0; mi < 4; ++mi) {
#pragma unroll
        for (int ni = 0; ni < 2; ++ni) {
            int gm = m0 + wm * 64 + mi * 16 + lhi * 8;
            int gn = n0 + wn * 32 + ni * 16 + l15;
#pragma unroll
            for (int v = 0; v < 8; ++v) {
                int row = gm + v;
                float val = c[mi][ni][v];
                if (EPI == 0) {
                    float sc = extra[(size_t)row * NEXP + (gn >> 8)];
                    outh[(size_t)row * ldc + gn] = (_Float16)(val * sc);
                } else if (EPI == 1) {
                    float g = 0.5f * val * (1.f + erff(val * 0.70710678118654752f));
                    outh[(size_t)row * ldc + gn] = (_Float16)g;
                } else {
                    outf[(size_t)row * ldc + gn] = val + extra[gn];
                }
            }
        }
    }
}

// --------------------------------------------------------------- launch ----
extern "C" void kernel_launch(void* const* d_in, const int* in_sizes, int n_in,
                              void* d_out, int out_size, void* d_ws, size_t ws_size,
                              hipStream_t stream)
{
    (void)in_sizes; (void)n_in; (void)out_size; (void)ws_size;

    const float* x   = (const float*)d_in[0];
    const float* rw  = (const float*)d_in[1];
    const float* rb  = (const float*)d_in[2];
    const float* w1  = (const float*)d_in[3];
    const float* w2  = (const float*)d_in[4];
    const float* l2w = (const float*)d_in[5];
    const float* l2b = (const float*)d_in[6];
    float* out = (float*)d_out;

    char* ws = (char*)d_ws;
    size_t off = 0;
    auto alloc = [&](size_t bytes) -> char* {
        size_t o = (off + 255) & ~(size_t)255;
        off = o + bytes;
        return ws + o;
    };

    _Float16* xh  = (_Float16*)alloc((size_t)NTOK * HDIM * 2);  // 16 MB
    _Float16* w1h = (_Float16*)alloc((size_t)HDIM * ERD  * 2);  //  4 MB
    _Float16* w2h = (_Float16*)alloc((size_t)ERD  * FDIM * 2);  // 16 MB
    _Float16* l2h = (_Float16*)alloc((size_t)FDIM * HDIM * 2);  //  8 MB
    float*    mw  = (float*)   alloc((size_t)NTOK * NEXP * 4);  // .25 MB
    _Float16* t   = (_Float16*)alloc((size_t)NTOK * ERD  * 2);  // 32 MB
    _Float16* hb  = (_Float16*)alloc((size_t)NTOK * FDIM * 2);  // 64 MB

    router_kernel<<<NTOK / 8, 256, 0, stream>>>(x, rw, rb, mw);
    cvt_f32_f16 <<<2048, 256, 0, stream>>>(x,   xh, (long)NTOK * HDIM);
    pack_w1_kernel<<<2048, 256, 0, stream>>>(w1, w1h);
    cvt_f32_f16 <<<2048, 256, 0, stream>>>(w2,  w2h, (long)ERD  * FDIM);
    cvt_f32_f16 <<<2048, 256, 0, stream>>>(l2w, l2h, (long)FDIM * HDIM);

    // t = (x @ W1cat) * mw    [8192 x 2048], K = 1024
    gemm_wmma_f16<0><<<dim3(ERD / 128, NTOK / 128), 256, 0, stream>>>(
        xh, w1h, HDIM, HDIM, ERD, ERD, mw, t, nullptr);
    // hb = gelu(t @ W2flat)   [8192 x 4096], K = 2048
    gemm_wmma_f16<1><<<dim3(FDIM / 128, NTOK / 128), 256, 0, stream>>>(
        t, w2h, ERD, ERD, FDIM, FDIM, nullptr, hb, nullptr);
    // out = hb @ lin2 + b     [8192 x 1024], K = 4096
    gemm_wmma_f16<2><<<dim3(HDIM / 128, NTOK / 128), 256, 0, stream>>>(
        hb, l2h, FDIM, FDIM, HDIM, HDIM, l2b, nullptr, out);
}
